// TriangleMultiplicationIncoming_72249939853678
// MI455X (gfx1250) — matrix operand order repaired and checked by reading, not compiled, not run on hardware
//
#include <hip/hip_runtime.h>
#include <hip/hip_bf16.h>
#include <stdint.h>

// ---------- types ----------
typedef __attribute__((ext_vector_type(16))) __bf16 v16bf;
typedef __attribute__((ext_vector_type(8)))  float  v8f;

struct alignas(16) U4 { unsigned int x, y, z, w; };

union Frag {
    U4           q[2];
    unsigned int u[8];
    v16bf        v;
};

#define S 512
#define C 128
#define NPOS (S * S)          // 262144 positions
#define PLANE ((size_t)NPOS)  // elements per channel plane

__device__ __forceinline__ unsigned short f2bf(float f) {
    unsigned int u = __builtin_bit_cast(unsigned int, f);
    u += 0x7FFFu + ((u >> 16) & 1u);   // round-to-nearest-even
    return (unsigned short)(u >> 16);
}
__device__ __forceinline__ float bf2f(unsigned short h) {
    unsigned int u = ((unsigned int)h) << 16;
    return __builtin_bit_cast(float, u);
}
// fast sigmoid: v_exp_f32 + v_rcp_f32 (avoid IEEE divide expansion)
__device__ __forceinline__ float sigmoidf(float x) {
    return __builtin_amdgcn_rcpf(1.0f + __expf(-x));
}
__device__ __forceinline__ unsigned int pack2(float lo, float hi) {
    return (unsigned int)f2bf(lo) | ((unsigned int)f2bf(hi) << 16);
}

// ---------- kernel 0: convert the four 128x128 weights to bf16 ----------
__global__ void __launch_bounds__(256)
cvt_weights(const float* __restrict__ Wa, const float* __restrict__ Wb,
            const float* __restrict__ Wg, const float* __restrict__ Wf,
            unsigned short* __restrict__ wbf) {
    int i = blockIdx.x * 256 + threadIdx.x;   // 0..16383
    wbf[            i] = f2bf(Wa[i]);
    wbf[1 * 16384 + i] = f2bf(Wb[i]);
    wbf[2 * 16384 + i] = f2bf(Wg[i]);
    wbf[3 * 16384 + i] = f2bf(Wf[i]);
}

// ---------- kernel 1: LN1 + projections (a,b,g) via WMMA ----------
// block = 128 thr (4 waves), handles 16 positions.
// a -> abuf[c][i][k]  (bf16), b -> bbuf[c][j][k] (bf16, transposed plane),
// g -> gbuf[p][d] (bf16, natural layout)
__global__ void __launch_bounds__(128)
ln_proj_kernel(const float* __restrict__ z,
               const float* __restrict__ w1, const float* __restrict__ b1,
               const unsigned short* __restrict__ wbf,
               unsigned short* __restrict__ abuf,
               unsigned short* __restrict__ bbuf,
               unsigned short* __restrict__ gbuf) {
    __shared__ float          zt[16][C];
    __shared__ float          red[16][8][2];
    __shared__ float          stats[16][2];
    __shared__ unsigned short zn[16][C];

    const int    t  = threadIdx.x;
    const size_t p0 = (size_t)blockIdx.x * 16;

    // load 16x128 fp32 tile (coalesced)
    #pragma unroll
    for (int i = 0; i < 16; ++i) {
        int idx = t + i * 128;
        zt[idx >> 7][idx & 127] = z[p0 * C + idx];
    }
    __syncthreads();

    // LN1: 8 threads per position, 16 channels each
    {
        int m = t >> 3, s8 = t & 7;
        float s = 0.f, ss = 0.f;
        #pragma unroll
        for (int c = s8 * 16; c < s8 * 16 + 16; ++c) {
            float v = zt[m][c]; s += v; ss += v * v;
        }
        red[m][s8][0] = s; red[m][s8][1] = ss;
    }
    __syncthreads();
    if (t < 16) {
        float s = 0.f, ss = 0.f;
        #pragma unroll
        for (int i = 0; i < 8; ++i) { s += red[t][i][0]; ss += red[t][i][1]; }
        float mu  = s * (1.0f / C);
        float var = ss * (1.0f / C) - mu * mu;
        stats[t][0] = mu;
        stats[t][1] = rsqrtf(var + 1e-5f);
    }
    __syncthreads();
    {
        int m = t >> 3, s8 = t & 7;
        float mu = stats[m][0], rs = stats[m][1];
        #pragma unroll
        for (int c = s8 * 16; c < s8 * 16 + 16; ++c)
            zn[m][c] = f2bf((zt[m][c] - mu) * rs * w1[c] + b1[c]);
    }
    __syncthreads();

    const int wave = t >> 5, lane = t & 31;
    const int ll = lane & 15, hi = lane >> 4;

    // hoist the 4 A fragments (16x32 each) out of the task loop
    Frag afr[4];
    #pragma unroll
    for (int kk = 0; kk < 4; ++kk) {
        int ka = kk * 32 + hi * 8;
        #pragma unroll
        for (int r = 0; r < 4; ++r) {
            afr[kk].u[r]     = *(const unsigned int*)&zn[ll][ka + 2 * r];
            afr[kk].u[4 + r] = *(const unsigned int*)&zn[ll][ka + 16 + 2 * r];
        }
    }

    // 3 weights x 8 n-tiles = 24 WMMA output tiles, 6 per wave
    for (int it = 0; it < 6; ++it) {
        const int task = wave * 6 + it;
        const int widx = __builtin_amdgcn_readfirstlane(task >> 3);  // 0=a 1=b 2=g
        const int nt   = __builtin_amdgcn_readfirstlane(task & 7);
        const int d    = nt * 16 + ll;

        // batch all four B fragments (8 x b128 in one clause), then chain WMMAs
        const unsigned short* wb = wbf + (size_t)widx * 16384 + (size_t)d * C;
        Frag bfr[4];
        #pragma unroll
        for (int kk = 0; kk < 4; ++kk) {
            const U4* pb = (const U4*)(wb + kk * 32 + hi * 8);
            bfr[kk].q[0] = pb[0]; bfr[kk].q[1] = pb[2];
        }
        v8f acc = {};
        #pragma unroll
        for (int kk = 0; kk < 4; ++kk)
            acc = __builtin_amdgcn_wmma_f32_16x16x32_bf16(
                false, afr[kk].v, false, bfr[kk].v, (short)0, acc, false, false);

        if (widx == 0) {
            // 8 consecutive p per lane -> one 16B store into [c][i][k]
            U4 pk;
            pk.x = pack2(sigmoidf(acc[0]), sigmoidf(acc[1]));
            pk.y = pack2(sigmoidf(acc[2]), sigmoidf(acc[3]));
            pk.z = pack2(sigmoidf(acc[4]), sigmoidf(acc[5]));
            pk.w = pack2(sigmoidf(acc[6]), sigmoidf(acc[7]));
            *(U4*)(abuf + (size_t)d * PLANE + p0 + (size_t)(hi * 8)) = pk;
        } else if (widx == 1) {
            // p=(k,j): store transposed plane [c][j][k], stride S per r
            size_t jbase = (p0 & 511) + (size_t)(hi * 8);
            size_t kcol  = p0 >> 9;
            unsigned short* dst = bbuf + (size_t)d * PLANE + jbase * S + kcol;
            #pragma unroll
            for (int r = 0; r < 8; ++r)
                dst[(size_t)r * S] = f2bf(sigmoidf(acc[r]));
        } else {
            // natural layout [p][d], stride C per r
            unsigned short* dst = gbuf + (p0 + (size_t)(hi * 8)) * C + d;
            #pragma unroll
            for (int r = 0; r < 8; ++r)
                dst[(size_t)r * C] = f2bf(sigmoidf(acc[r]));
        }
    }
}

// ---------- kernel 2: per-channel 512x512x512 batched GEMM (bf16 WMMA) ----------
// grid (4,4,128), 256 thr (8 waves). Block tile 128x128; wave tile 32x64.
// Register double-buffered K loop (unrolled x2 so buffer indices are constant).
__global__ void __launch_bounds__(256)
chan_gemm_kernel(const unsigned short* __restrict__ abuf,
                 const unsigned short* __restrict__ bbuf,
                 float* __restrict__ sumbuf) {
    const int c  = blockIdx.z;
    const int jb = blockIdx.x * 128, ib = blockIdx.y * 128;
    const int t = threadIdx.x, wave = t >> 5, lane = t & 31;
    const int ll = lane & 15, hi = lane >> 4;
    const int i0 = ib + (wave >> 1) * 32;
    const int j0 = jb + (wave & 1) * 64;

    const unsigned short* ap = abuf + (size_t)c * PLANE;  // rows i, contiguous k
    const unsigned short* bp = bbuf + (size_t)c * PLANE;  // rows j, contiguous k

    // per-lane K-contiguous row base pointers
    const unsigned short* parow0 = ap + (size_t)(i0 + ll) * S + hi * 8;
    const unsigned short* parow1 = parow0 + (size_t)16 * S;
    const unsigned short* pbrow  = bp + (size_t)(j0 + ll) * S + hi * 8;

    v8f  acc[2][4] = {};
    Frag af[2][2], bf[2][4];

    auto load_step = [&](int buf, int k0) {
        const U4* pa0 = (const U4*)(parow0 + k0);
        const U4* pa1 = (const U4*)(parow1 + k0);
        af[buf][0].q[0] = pa0[0]; af[buf][0].q[1] = pa0[2];   // K +0..7, +16..23
        af[buf][1].q[0] = pa1[0]; af[buf][1].q[1] = pa1[2];
        #pragma unroll
        for (int nt = 0; nt < 4; ++nt) {
            const U4* pb = (const U4*)(pbrow + (size_t)(nt * 16) * S + k0);
            bf[buf][nt].q[0] = pb[0]; bf[buf][nt].q[1] = pb[2]; // K +0..7, +16..23
        }
    };
    auto wmma_step = [&](int buf) {
        #pragma unroll
        for (int mt = 0; mt < 2; ++mt)
            #pragma unroll
            for (int nt = 0; nt < 4; ++nt)
                acc[mt][nt] = __builtin_amdgcn_wmma_f32_16x16x32_bf16(
                    false, af[buf][mt].v, false, bf[buf][nt].v, (short)0,
                    acc[mt][nt], false, false);
    };

    load_step(0, 0);
    #pragma unroll 1
    for (int k0 = 0; k0 < S - 64; k0 += 64) {
        load_step(1, k0 + 32);   // prefetch while wmma(0) runs
        wmma_step(0);
        load_step(0, k0 + 64);   // prefetch while wmma(1) runs
        wmma_step(1);
    }
    load_step(1, S - 32);        // tail: k0 = 448 already in buf 0
    wmma_step(0);
    wmma_step(1);

    #pragma unroll
    for (int mt = 0; mt < 2; ++mt)
        #pragma unroll
        for (int nt = 0; nt < 4; ++nt)
            #pragma unroll
            for (int r = 0; r < 8; ++r) {
                int mi = i0 + mt * 16 + r + hi * 8;
                int nj = j0 + nt * 16 + ll;
                sumbuf[(size_t)c * PLANE + (size_t)mi * S + nj] = acc[mt][nt][r];
            }
}

// ---------- kernel 3: LN2 + Wf projection + gate ----------
__global__ void __launch_bounds__(128)
ln2_out_kernel(const float* __restrict__ sumbuf,
               const float* __restrict__ w2, const float* __restrict__ b2,
               const unsigned short* __restrict__ wfbf,
               const unsigned short* __restrict__ gbuf,
               float* __restrict__ out) {
    __shared__ float          st[16][C];
    __shared__ float          red[16][8][2];
    __shared__ float          stats[16][2];
    __shared__ unsigned short zn[16][C];

    const int    t  = threadIdx.x;
    const size_t p0 = (size_t)blockIdx.x * 16;

    // gather 16 positions x 128 channels from channel-major sum planes
    #pragma unroll
    for (int i = 0; i < 16; ++i) {
        int idx = t + i * 128;       // 0..2047
        int cc = idx >> 4, m = idx & 15;
        st[m][cc] = sumbuf[(size_t)cc * PLANE + p0 + m];
    }
    __syncthreads();

    {
        int m = t >> 3, s8 = t & 7;
        float s = 0.f, ss = 0.f;
        #pragma unroll
        for (int cc = s8 * 16; cc < s8 * 16 + 16; ++cc) {
            float v = st[m][cc]; s += v; ss += v * v;
        }
        red[m][s8][0] = s; red[m][s8][1] = ss;
    }
    __syncthreads();
    if (t < 16) {
        float s = 0.f, ss = 0.f;
        #pragma unroll
        for (int i = 0; i < 8; ++i) { s += red[t][i][0]; ss += red[t][i][1]; }
        float mu  = s * (1.0f / C);
        float var = ss * (1.0f / C) - mu * mu;
        stats[t][0] = mu;
        stats[t][1] = rsqrtf(var + 1e-5f);
    }
    __syncthreads();
    {
        int m = t >> 3, s8 = t & 7;
        float mu = stats[m][0], rs = stats[m][1];
        #pragma unroll
        for (int cc = s8 * 16; cc < s8 * 16 + 16; ++cc)
            zn[m][cc] = f2bf((st[m][cc] - mu) * rs * w2[cc] + b2[cc]);
    }
    __syncthreads();

    const int wave = t >> 5, lane = t & 31;
    const int ll = lane & 15, hi = lane >> 4;

    Frag afr[4];
    #pragma unroll
    for (int kk = 0; kk < 4; ++kk) {
        int ka = kk * 32 + hi * 8;
        #pragma unroll
        for (int r = 0; r < 4; ++r) {
            afr[kk].u[r]     = *(const unsigned int*)&zn[ll][ka + 2 * r];
            afr[kk].u[4 + r] = *(const unsigned int*)&zn[ll][ka + 16 + 2 * r];
        }
    }

    for (int it = 0; it < 2; ++it) {
        const int nt = __builtin_amdgcn_readfirstlane(wave * 2 + it);
        const int d  = nt * 16 + ll;

        const unsigned short* wb = wfbf + (size_t)d * C;
        Frag bfr[4];
        #pragma unroll
        for (int kk = 0; kk < 4; ++kk) {
            const U4* pb = (const U4*)(wb + kk * 32 + hi * 8);
            bfr[kk].q[0] = pb[0]; bfr[kk].q[1] = pb[2];
        }
        v8f acc = {};
        #pragma unroll
        for (int kk = 0; kk < 4; ++kk)
            acc = __builtin_amdgcn_wmma_f32_16x16x32_bf16(
                false, afr[kk].v, false, bfr[kk].v, (short)0, acc, false, false);

        #pragma unroll
        for (int r = 0; r < 8; ++r) {
            size_t p = p0 + (size_t)(r + hi * 8);
            float g = bf2f(gbuf[p * C + d]);
            out[p * C + d] = g * acc[r];   // lanes 0-15 -> consecutive d
        }
    }
}

// ---------- launcher ----------
extern "C" void kernel_launch(void* const* d_in, const int* in_sizes, int n_in,
                              void* d_out, int out_size, void* d_ws, size_t ws_size,
                              hipStream_t stream) {
    const float* z     = (const float*)d_in[0];
    const float* ln1_w = (const float*)d_in[1];
    const float* ln1_b = (const float*)d_in[2];
    const float* Wa    = (const float*)d_in[3];
    const float* Wb    = (const float*)d_in[4];
    const float* Wg    = (const float*)d_in[5];
    const float* ln2_w = (const float*)d_in[6];
    const float* ln2_b = (const float*)d_in[7];
    const float* Wf    = (const float*)d_in[8];
    float* out = (float*)d_out;

    // workspace layout (bytes):
    //   wbf   : 4*128*128 bf16          = 131072
    //   abuf  : 128*262144 bf16         = 67108864   [c][i][k]
    //   bbuf  : 128*262144 bf16         = 67108864   [c][j][k]
    //   gbuf  : 262144*128 bf16         = 67108864   [p][d]
    //   sumbuf: 128*262144 f32          = 134217728  [c][i][j]
    char* ws = (char*)d_ws;
    unsigned short* wbf  = (unsigned short*)ws;
    unsigned short* abuf = (unsigned short*)(ws + 131072);
    unsigned short* bbuf = abuf + (size_t)C * PLANE;
    unsigned short* gbuf = bbuf + (size_t)C * PLANE;
    float*          sumbuf = (float*)(gbuf + (size_t)C * PLANE);

    cvt_weights<<<64, 256, 0, stream>>>(Wa, Wb, Wg, Wf, wbf);
    ln_proj_kernel<<<NPOS / 16, 128, 0, stream>>>(z, ln1_w, ln1_b, wbf,
                                                  abuf, bbuf, gbuf);
    chan_gemm_kernel<<<dim3(4, 4, 128), 256, 0, stream>>>(abuf, bbuf, sumbuf);
    ln2_out_kernel<<<NPOS / 16, 128, 0, stream>>>(sumbuf, ln2_w, ln2_b,
                                                  wbf + 3 * 16384, gbuf, out);
}
